// ActorTransformer_91113436218022
// MI455X (gfx1250) — compile-verified, matrix operand order checked by reading, not run on hardware
//
#include <hip/hip_runtime.h>
#include <hip/hip_bf16.h>

// ---------------- problem constants (from reference) ----------------
#define LYR 6
#define NHD 6
#define DIM 96
#define HS  16
#define FFD 384
#define DG  32
#define DO  64
#define DA  16
#define TT  1024
#define BB  16
#define BT  (BB * TT)   // 16384

typedef __bf16 bf16;
typedef __attribute__((ext_vector_type(16))) __bf16 v16bf;
typedef __attribute__((ext_vector_type(8)))  __bf16 v8bf;
typedef __attribute__((ext_vector_type(8)))  float  v8f;

__device__ __forceinline__ v16bf cat16(v8bf lo, v8bf hi) {
    return __builtin_shufflevector(lo, hi, 0,1,2,3,4,5,6,7,8,9,10,11,12,13,14,15);
}

__device__ __forceinline__ v8f wmma_bf16(v16bf a, v16bf b, v8f c) {
    // D = A(16x32 bf16) * B(32x16 bf16) + C(16x16 f32)
    return __builtin_amdgcn_wmma_f32_16x16x32_bf16(false, a, false, b, (short)0, c, false, false);
}

// ---------------- weight conversion: fp32 [K][N] -> bf16 [batch][N][K] ----------------
__global__ void conv_tn_kernel(const float* __restrict__ src, bf16* __restrict__ dst,
                               int N, int K, long total) {
    long i = (long)blockIdx.x * blockDim.x + threadIdx.x;
    if (i >= total) return;
    int  k  = (int)(i % K);
    long r  = i / K;
    int  n  = (int)(r % N);
    long bi = r / N;
    dst[i] = (bf16)src[bi * (long)N * K + (long)k * N + n];
}

// wq/wk/wv [L][NH][D][HS] -> bf16 [L][ (h*HS+j) ][ d ]  (i.e. per-layer W^T [96][96])
__global__ void conv_qkv_kernel(const float* __restrict__ wq, const float* __restrict__ wk,
                                const float* __restrict__ wv,
                                bf16* __restrict__ WqT, bf16* __restrict__ WkT,
                                bf16* __restrict__ WvT) {
    long i = (long)blockIdx.x * blockDim.x + threadIdx.x;
    const long total = (long)LYR * NHD * DIM * HS;
    if (i >= total) return;
    int  j = (int)(i % HS);
    long r = i / HS;
    int  d = (int)(r % DIM); r /= DIM;
    int  h = (int)(r % NHD);
    int  l = (int)(r / NHD);
    long dst = ((long)l * DIM + (h * HS + j)) * DIM + d;
    WqT[dst] = (bf16)wq[i];
    WkT[dst] = (bf16)wk[i];
    WvT[dst] = (bf16)wv[i];
}

// ---------------- concat(goal, obs) -> bf16 [BT][96] ----------------
__global__ void concat_kernel(const float* __restrict__ goals, const float* __restrict__ obss,
                              bf16* __restrict__ Xc) {
    long i = (long)blockIdx.x * blockDim.x + threadIdx.x;
    const long total = (long)BT * DIM;
    if (i >= total) return;
    int  c  = (int)(i % DIM);
    long bt = i / DIM;
    int  b  = (int)(bt / TT);
    int  t  = (int)(bt % TT);
    float v = (c < DG) ? goals[(long)b * DG + c]
                       : obss[((long)b * TT + t) * DO + (c - DG)];
    Xc[i] = (bf16)v;
}

// ---------------- layernorm (f32 in -> bf16 out), one wave per row of 96 ----------------
__global__ void ln_kernel(const float* __restrict__ x, const float* __restrict__ g,
                          const float* __restrict__ bta, bf16* __restrict__ out, int rows) {
    int wave = threadIdx.x >> 5, lane = threadIdx.x & 31;
    int row  = blockIdx.x * (blockDim.x >> 5) + wave;
    if (row >= rows) return;
    const float* xr = x + (long)row * DIM;
    float a0 = xr[lane], a1 = xr[lane + 32], a2 = xr[lane + 64];
    float s = a0 + a1 + a2;
    float q = a0 * a0 + a1 * a1 + a2 * a2;
#pragma unroll
    for (int off = 1; off < 32; off <<= 1) {
        s += __shfl_xor(s, off);
        q += __shfl_xor(q, off);
    }
    float mean = s * (1.0f / DIM);
    float var  = q * (1.0f / DIM) - mean * mean;
    float rstd = rsqrtf(var + 1e-5f);
    bf16* orow = out + (long)row * DIM;
    orow[lane]      = (bf16)((a0 - mean) * rstd * g[lane]      + bta[lane]);
    orow[lane + 32] = (bf16)((a1 - mean) * rstd * g[lane + 32] + bta[lane + 32]);
    orow[lane + 64] = (bf16)((a2 - mean) * rstd * g[lane + 64] + bta[lane + 64]);
}

// ---------------- generic WMMA GEMM: C[M,N] = A_bf16[M,K] @ Bt_bf16[N,K]^T ----------------
// epilogue: (+bias[n]) (+resid[(m % res_mod)*N+n]) (relu) -> f32 or bf16
__global__ void gemm_wmma_kernel(const bf16* __restrict__ A, const bf16* __restrict__ Bt,
                                 float* outF, bf16* outB,
                                 const float* __restrict__ bias,
                                 const float* __restrict__ resid, int res_mod,
                                 int M, int N, int K, int relu) {
    int lane = threadIdx.x & 31;
    int wave = threadIdx.x >> 5;
    int m0 = (blockIdx.x * 4 + wave) * 16;
    int n0 = blockIdx.y * 16;
    if (m0 >= M) return;
    int r     = lane & 15;
    int base0 = (lane < 16) ? 0 : 8;
    const bf16* Ap = A  + (long)(m0 + r) * K + base0;
    const bf16* Bp = Bt + (long)(n0 + r) * K + base0;
    v8f acc = {};
    for (int k0 = 0; k0 < K; k0 += 32) {
        v8bf alo = *(const v8bf*)(Ap + k0);
        v8bf ahi = *(const v8bf*)(Ap + k0 + 16);
        v8bf blo = *(const v8bf*)(Bp + k0);
        v8bf bhi = *(const v8bf*)(Bp + k0 + 16);
        acc = wmma_bf16(cat16(alo, ahi), cat16(blo, bhi), acc);
    }
    int   n  = n0 + r;
    float bv = bias ? bias[n] : 0.0f;
#pragma unroll
    for (int v = 0; v < 8; ++v) {
        int m = m0 + ((lane < 16) ? v : v + 8);
        float val = acc[v] + bv;
        if (resid) {
            int rr = res_mod ? (m % res_mod) : m;
            val += resid[(long)rr * N + n];
        }
        if (relu) val = fmaxf(val, 0.0f);
        long idx = (long)m * N + n;
        if (outF) outF[idx] = val;
        else      outB[idx] = (bf16)val;
    }
}

// ---------------- V transpose: Vb[BT][96] -> Vt[B][NH][HS][T] (bf16) ----------------
__global__ void transpose_v_kernel(const bf16* __restrict__ Vb, bf16* __restrict__ Vt) {
    long i = (long)blockIdx.x * blockDim.x + threadIdx.x;
    const long total = (long)BB * NHD * HS * TT;
    if (i >= total) return;
    int  t = (int)(i % TT);
    long r = i / TT;
    int  j = (int)(r % HS); r /= HS;
    int  h = (int)(r % NHD);
    int  b = (int)(r / NHD);
    Vt[i] = Vb[((long)b * TT + t) * DIM + h * HS + j];
}

// ---------------- flash attention: 1 wave per (b, h, 16-query tile), 32-key chunks ----------------
__global__ void attn_kernel(const bf16* __restrict__ Qb, const bf16* __restrict__ Kb,
                            const bf16* __restrict__ Vt, bf16* __restrict__ Ob) {
    __shared__ __align__(16) bf16 P[16][32];   // P tile, dense 16x32 bf16 (A layout source)
    int lane = threadIdx.x;
    int q0   = blockIdx.x * 16;
    int bh   = blockIdx.y;
    int b = bh / NHD, h = bh % NHD;
    int r     = lane & 15;
    int base0 = (lane < 16) ? 0 : 8;
    v8bf z8 = {};

    // Q tile in A layout: row = q0+r, head dims base0..base0+7 ; K-dim padded 16->32 with zeros
    const bf16* Qp = Qb + ((long)(b * TT + q0 + r)) * DIM + h * HS + base0;
    v16bf aQ = cat16(*(const v8bf*)Qp, z8);

    const bf16* Kbase = Kb + ((long)b * TT + r) * DIM + h * HS + base0;   // + (s0)*DIM steps
    const bf16* Vp    = Vt + ((long)bh * HS + r) * TT + base0;            // row = head dim r, K = key idx

    v8f   accO = {};
    float mrun[8], lrun[8];
#pragma unroll
    for (int v = 0; v < 8; ++v) { mrun[v] = -1e30f; lrun[v] = 0.0f; }

    for (int s0 = 0; s0 <= q0; s0 += 32) {
        // ---- scores S[16, 0..31] = Q @ K^T : two 16-key B tiles from K rows s0+r, s0+16+r
        const bf16* Kp = Kbase + (long)s0 * DIM;
        if (s0 + 32 <= q0)  // prefetch next chunk's K rows (global_prefetch)
            __builtin_prefetch(Kp + 32 * DIM, 0, 1);
        v16bf bK0 = cat16(*(const v8bf*)Kp, z8);
        v16bf bK1 = cat16(*(const v8bf*)(Kp + 16 * DIM), z8);
        v8f s_lo = {}, s_hi = {};
        s_lo = wmma_bf16(aQ, bK0, s_lo);
        s_hi = wmma_bf16(aQ, bK1, s_hi);

        int  n    = r;                        // key index within chunk (C layout column)
        bool diag = (s0 + 31 > q0);           // chunk straddles the causal diagonal
#pragma unroll
        for (int v = 0; v < 8; ++v) {
            int m = (lane < 16) ? v : v + 8;  // query row within tile
            float sc0 = s_lo[v] * 0.25f;      // HS^-0.5
            float sc1 = s_hi[v] * 0.25f;
            if (diag) {
                if (s0 + n      > q0 + m) sc0 = -1e30f;
                if (s0 + n + 16 > q0 + m) sc1 = -1e30f;
            }
            // row max over 32 keys: per-lane max, then reduce over the 16 lanes of this half
            float rm = fmaxf(sc0, sc1);
            rm = fmaxf(rm, __shfl_xor(rm, 1));
            rm = fmaxf(rm, __shfl_xor(rm, 2));
            rm = fmaxf(rm, __shfl_xor(rm, 4));
            rm = fmaxf(rm, __shfl_xor(rm, 8));
            float mnew = fmaxf(mrun[v], rm);
            float p0   = __expf(sc0 - mnew);
            float p1   = __expf(sc1 - mnew);
            float rs   = p0 + p1;
            rs += __shfl_xor(rs, 1);
            rs += __shfl_xor(rs, 2);
            rs += __shfl_xor(rs, 4);
            rs += __shfl_xor(rs, 8);
            float alpha = __expf(mrun[v] - mnew);
            lrun[v] = lrun[v] * alpha + rs;
            mrun[v] = mnew;
            accO[v] *= alpha;
            P[m][n]      = (bf16)p0;          // stage P for C->A layout conversion
            P[m][n + 16] = (bf16)p1;
        }
        __syncthreads();
        // ---- O += P(16x32) @ V(32x16), full K=32, no padding
        v16bf aP = cat16(*(const v8bf*)(&P[r][base0]),
                         *(const v8bf*)(&P[r][base0 + 16]));
        v16bf bV = cat16(*(const v8bf*)(Vp + s0),
                         *(const v8bf*)(Vp + s0 + 16));
        accO = wmma_bf16(aP, bV, accO);
        __syncthreads();
    }

#pragma unroll
    for (int v = 0; v < 8; ++v) {
        int m = (lane < 16) ? v : v + 8;
        float val = accO[v] / lrun[v];
        Ob[((long)(b * TT + q0 + m)) * DIM + h * HS + r] = (bf16)val;
    }
}

// ---------------- host orchestration ----------------
static inline long cdiv(long a, long b) { return (a + b - 1) / b; }

extern "C" void kernel_launch(void* const* d_in, const int* in_sizes, int n_in,
                              void* d_out, int out_size, void* d_ws, size_t ws_size,
                              hipStream_t stream) {
    const float* goals  = (const float*)d_in[0];
    const float* obss   = (const float*)d_in[1];
    const float* w_go   = (const float*)d_in[2];
    const float* b_go   = (const float*)d_in[3];
    const float* pos    = (const float*)d_in[4];
    const float* wq     = (const float*)d_in[5];
    const float* wk     = (const float*)d_in[6];
    const float* wv     = (const float*)d_in[7];
    const float* w_proj = (const float*)d_in[8];
    const float* b_proj = (const float*)d_in[9];
    const float* ln1_g  = (const float*)d_in[10];
    const float* ln1_b  = (const float*)d_in[11];
    const float* ln2_g  = (const float*)d_in[12];
    const float* ln2_b  = (const float*)d_in[13];
    const float* w_ff1  = (const float*)d_in[14];
    const float* b_ff1  = (const float*)d_in[15];
    const float* w_ff2  = (const float*)d_in[16];
    const float* b_ff2  = (const float*)d_in[17];
    const float* lnf_g  = (const float*)d_in[18];
    const float* lnf_b  = (const float*)d_in[19];
    const float* w_act  = (const float*)d_in[20];
    const float* b_act  = (const float*)d_in[21];
    float* out = (float*)d_out;

    // workspace carve-up (256B aligned slices)
    char*  base = (char*)d_ws;
    size_t off  = 0;
    auto alloc = [&](size_t bytes) -> char* {
        char* p = base + off;
        off += (bytes + 255) & ~(size_t)255;
        return p;
    };
    bf16*  WgoT   = (bf16*)alloc((size_t)DIM * DIM * 2);
    bf16*  WqT    = (bf16*)alloc((size_t)LYR * DIM * DIM * 2);
    bf16*  WkT    = (bf16*)alloc((size_t)LYR * DIM * DIM * 2);
    bf16*  WvT    = (bf16*)alloc((size_t)LYR * DIM * DIM * 2);
    bf16*  WprojT = (bf16*)alloc((size_t)LYR * DIM * DIM * 2);
    bf16*  Wff1T  = (bf16*)alloc((size_t)LYR * FFD * DIM * 2);
    bf16*  Wff2T  = (bf16*)alloc((size_t)LYR * DIM * FFD * 2);
    bf16*  WactT  = (bf16*)alloc((size_t)DA * DIM * 2);
    bf16*  Xc     = (bf16*)alloc((size_t)BT * DIM * 2);
    float* x      = (float*)alloc((size_t)BT * DIM * 4);
    bf16*  xn     = (bf16*)alloc((size_t)BT * DIM * 2);
    bf16*  Qb     = (bf16*)alloc((size_t)BT * DIM * 2);
    bf16*  Kbuf   = (bf16*)alloc((size_t)BT * DIM * 2);
    bf16*  Vb     = (bf16*)alloc((size_t)BT * DIM * 2);
    bf16*  Vt     = (bf16*)alloc((size_t)BB * NHD * HS * TT * 2);
    bf16*  Ob     = (bf16*)alloc((size_t)BT * DIM * 2);
    bf16*  Hb     = (bf16*)alloc((size_t)BT * FFD * 2);
    (void)ws_size;

    const int TPB = 256;

    // --- weight conversion (bf16, transposed to [N][K]) ---
    {
        long t;
        t = (long)DIM * DIM;
        conv_tn_kernel<<<cdiv(t, TPB), TPB, 0, stream>>>(w_go, WgoT, DIM, DIM, t);
        t = (long)LYR * NHD * DIM * HS;
        conv_qkv_kernel<<<cdiv(t, TPB), TPB, 0, stream>>>(wq, wk, wv, WqT, WkT, WvT);
        t = (long)LYR * DIM * DIM;
        conv_tn_kernel<<<cdiv(t, TPB), TPB, 0, stream>>>(w_proj, WprojT, DIM, DIM, t);
        t = (long)LYR * FFD * DIM;
        conv_tn_kernel<<<cdiv(t, TPB), TPB, 0, stream>>>(w_ff1, Wff1T, FFD, DIM, t);
        t = (long)LYR * DIM * FFD;
        conv_tn_kernel<<<cdiv(t, TPB), TPB, 0, stream>>>(w_ff2, Wff2T, DIM, FFD, t);
        t = (long)DA * DIM;
        conv_tn_kernel<<<cdiv(t, TPB), TPB, 0, stream>>>(w_act, WactT, DA, DIM, t);
    }

    // --- embedding: x = concat(goal,obs) @ w_go + b_go + pos_emb ---
    {
        long t = (long)BT * DIM;
        concat_kernel<<<cdiv(t, TPB), TPB, 0, stream>>>(goals, obss, Xc);
        dim3 g(BT / 64, DIM / 16);
        gemm_wmma_kernel<<<g, 128, 0, stream>>>(Xc, WgoT, x, nullptr, b_go,
                                                pos, TT, BT, DIM, DIM, 0);
    }

    dim3 g96(BT / 64, DIM / 16);
    dim3 gff(BT / 64, FFD / 16);
    dim3 gattn(TT / 16, BB * NHD);

    for (int l = 0; l < LYR; ++l) {
        // ln1 -> xn (bf16)
        ln_kernel<<<BT / 4, 128, 0, stream>>>(x, ln1_g + l * DIM, ln1_b + l * DIM, xn, BT);
        // Q, K, V projections (heads concatenated -> N=96)
        gemm_wmma_kernel<<<g96, 128, 0, stream>>>(xn, WqT + (long)l * DIM * DIM, nullptr, Qb,
                                                  nullptr, nullptr, 0, BT, DIM, DIM, 0);
        gemm_wmma_kernel<<<g96, 128, 0, stream>>>(xn, WkT + (long)l * DIM * DIM, nullptr, Kbuf,
                                                  nullptr, nullptr, 0, BT, DIM, DIM, 0);
        gemm_wmma_kernel<<<g96, 128, 0, stream>>>(xn, WvT + (long)l * DIM * DIM, nullptr, Vb,
                                                  nullptr, nullptr, 0, BT, DIM, DIM, 0);
        // V -> Vt [B][NH][HS][T]
        {
            long t = (long)BB * NHD * HS * TT;
            transpose_v_kernel<<<cdiv(t, TPB), TPB, 0, stream>>>(Vb, Vt);
        }
        // causal flash attention -> Ob (bf16)
        attn_kernel<<<gattn, 32, 0, stream>>>(Qb, Kbuf, Vt, Ob);
        // x = x + Ob @ w_proj + b_proj
        gemm_wmma_kernel<<<g96, 128, 0, stream>>>(Ob, WprojT + (long)l * DIM * DIM, x, nullptr,
                                                  b_proj + l * DIM, x, 0, BT, DIM, DIM, 0);
        // ln2 -> xn
        ln_kernel<<<BT / 4, 128, 0, stream>>>(x, ln2_g + l * DIM, ln2_b + l * DIM, xn, BT);
        // Hb = relu(xn @ w_ff1 + b_ff1)  (bf16)
        gemm_wmma_kernel<<<gff, 128, 0, stream>>>(xn, Wff1T + (long)l * FFD * DIM, nullptr, Hb,
                                                  b_ff1 + l * FFD, nullptr, 0, BT, FFD, DIM, 1);
        // x = x + Hb @ w_ff2 + b_ff2
        gemm_wmma_kernel<<<g96, 128, 0, stream>>>(Hb, Wff2T + (long)l * DIM * FFD, x, nullptr,
                                                  b_ff2 + l * DIM, x, 0, BT, DIM, FFD, 0);
    }

    // final layernorm + act head
    ln_kernel<<<BT / 4, 128, 0, stream>>>(x, lnf_g, lnf_b, xn, BT);
    {
        dim3 g(BT / 64, DA / 16);
        gemm_wmma_kernel<<<g, 128, 0, stream>>>(xn, WactT, out, nullptr, b_act,
                                                nullptr, 0, BT, DA, DIM, 0);
    }
}
